// VectorQuantizer_541165879472
// MI455X (gfx1250) — matrix operand order repaired and checked by reading, not compiled, not run on hardware
//
#include <hip/hip_runtime.h>
#include <hip/hip_bf16.h>
#include <math.h>

// ---------- problem constants (from reference) ----------
#define DDIM    256          // embedding dim
#define KCODES  2048         // codebook size
#define NROWS   65536        // 16*4096 flattened vectors
#define BETA    0.25f

// output layout: (z_q_st, loss, indices, perplexity) concatenated flat
#define OFF_ZQ    0
#define OFF_LOSS  ((size_t)NROWS * DDIM)                 // 16777216
#define OFF_IDX   (OFF_LOSS + 1)                         // 16777217
#define OFF_PERP  (OFF_IDX + NROWS)                      // 16842753

// ---------- tiling ----------
#define WAVES          8                    // waves per block
#define BLOCK_THREADS  (WAVES * 32)
#define ROWS_PER_WAVE  32                   // two 16-row A panels
#define ROWS_PER_BLOCK (WAVES * ROWS_PER_WAVE)   // 256
#define CHUNK          128                  // codes staged per LDS buffer
#define NCHUNK         (KCODES / CHUNK)     // 16
#define CHUNK_BYTES    (CHUNK * DDIM * 2)   // 64 KB bf16

typedef __attribute__((ext_vector_type(16))) __bf16 v16bf;
typedef __attribute__((ext_vector_type(4)))  __bf16 v4bf;
typedef __attribute__((ext_vector_type(8)))  float  v8f;

// Generic pointers into LDS carry the LDS byte offset in their low 32 bits
__device__ __forceinline__ unsigned lds_off(const void* p) {
    return (unsigned)(unsigned long long)p;
}

// Async DMA one 64KB codebook chunk into LDS (ASYNCcnt path, gfx1250).
// 256 threads x 16 x b128 = 64 KB; each instruction moves 512B contiguous.
__device__ __forceinline__ void prefetch_chunk(unsigned ldsBase,
                                               const char* gsrc, int tid) {
    const unsigned l = ldsBase + (unsigned)tid * 16u;
    const unsigned long long g = (unsigned long long)gsrc + (unsigned long long)tid * 16ull;
#pragma unroll
    for (int i = 0; i < CHUNK_BYTES / (BLOCK_THREADS * 16); ++i) {   // 16
        asm volatile("global_load_async_to_lds_b128 %0, %1, off"
                     :: "v"(l + (unsigned)(i * BLOCK_THREADS * 16)),
                        "v"(g + (unsigned long long)(i * BLOCK_THREADS * 16))
                     : "memory");
    }
}

__device__ __forceinline__ void wait_async0() {
    asm volatile("s_wait_asynccnt 0x0" ::: "memory");
}

// ---------------------------------------------------------------------------
// Kernel A: codebook f32 -> bf16, exact f32 row norms, zero accumulators.
// ---------------------------------------------------------------------------
__global__ __launch_bounds__(32) void vq_prep(const float* __restrict__ embed,
                                              __bf16* __restrict__ ebf,
                                              float* __restrict__ enorm,
                                              float* __restrict__ counts,
                                              float* __restrict__ lossAcc) {
    const int k    = blockIdx.x;
    const int lane = threadIdx.x;
    const float* er = embed + (size_t)k * DDIM;
    __bf16*      eb = ebf   + (size_t)k * DDIM;

    float s = 0.0f;
#pragma unroll
    for (int j = 0; j < 2; ++j) {
        const int dd = lane * 8 + j * 4;
        float4 f = *(const float4*)(er + dd);
        s += f.x * f.x + f.y * f.y + f.z * f.z + f.w * f.w;
        v4bf b;
        b[0] = (__bf16)f.x; b[1] = (__bf16)f.y;
        b[2] = (__bf16)f.z; b[3] = (__bf16)f.w;
        *(v4bf*)(eb + dd) = b;
    }
#pragma unroll
    for (int off = 1; off < 32; off <<= 1) s += __shfl_xor(s, off, 32);

    if (lane == 0) { enorm[k] = s; counts[k] = 0.0f; }
    if (k == 0 && lane == 1) *lossAcc = 0.0f;
}

// ---------------------------------------------------------------------------
// Kernel B: 8 waves x 32 rows per block. Register-resident bf16 A panels,
// codebook double-buffered in LDS via global_load_async_to_lds_b128, WMMA
// argmin on score = ||e||^2 - 2 z.e, then exact f32 gather/loss/histogram.
// Tile loop kept rolled (unroll 1) so the live set stays < 256 VGPRs:
// A panels 128 + best/bidx 32 + acc 16 + B staging ~32 -> no vgpr-msb churn,
// 4 waves/SIMD.
// ---------------------------------------------------------------------------
__global__ __launch_bounds__(BLOCK_THREADS)
void vq_main(const float* __restrict__ z,
             const float* __restrict__ embed,
             const __bf16* __restrict__ ebf,
             const float* __restrict__ enorm,
             float* __restrict__ out,
             float* __restrict__ counts,
             float* __restrict__ lossAcc) {
    const int tid    = threadIdx.x;
    const int lane   = tid & 31;
    const int waveId = tid >> 5;
    const int m      = lane & 15;       // row within A tile / col within B tile
    const int hi     = lane >> 4;       // lane half selects K sub-range
    const int waveRow = blockIdx.x * ROWS_PER_BLOCK + waveId * ROWS_PER_WAVE;

    __shared__ __bf16 sBuf[2][CHUNK * DDIM];    // 2 x 64 KB double buffer
    __shared__ float  sEnorm[KCODES];           // 8 KB
    __shared__ int    sIdx[WAVES][ROWS_PER_WAVE];

    // stage exact code norms once (published by first chunk barrier)
    for (int k = tid; k < KCODES; k += BLOCK_THREADS) sEnorm[k] = enorm[k];

    // ---- A panels: 32 rows x 256 cols, f32 -> bf16, resident in 128 VGPRs --
    // A in this code's contiguous k order (same as B): lane L holds row L%16; K-run of 16 at
    // d0 + 16*(L/16).
    v16bf a[2][8];
#pragma unroll
    for (int p = 0; p < 2; ++p) {
        const float* zrow = z + (size_t)(waveRow + p * 16 + m) * DDIM;
#pragma unroll 4
        for (int c = 0; c < 8; ++c) {
            const int d0 = c * 32 + hi * 16;
            float4 f0 = *(const float4*)(zrow + d0);
            float4 f1 = *(const float4*)(zrow + d0 + 4);
            float4 f2 = *(const float4*)(zrow + d0 + 8);
            float4 f3 = *(const float4*)(zrow + d0 + 12);
            v16bf av;
            av[0]  = (__bf16)f0.x; av[1]  = (__bf16)f0.y; av[2]  = (__bf16)f0.z; av[3]  = (__bf16)f0.w;
            av[4]  = (__bf16)f1.x; av[5]  = (__bf16)f1.y; av[6]  = (__bf16)f1.z; av[7]  = (__bf16)f1.w;
            av[8]  = (__bf16)f2.x; av[9]  = (__bf16)f2.y; av[10] = (__bf16)f2.z; av[11] = (__bf16)f2.w;
            av[12] = (__bf16)f3.x; av[13] = (__bf16)f3.y; av[14] = (__bf16)f3.z; av[15] = (__bf16)f3.w;
            a[p][c] = av;
        }
    }

    float best[2][8];
    int   bidx[2][8];
#pragma unroll
    for (int p = 0; p < 2; ++p)
#pragma unroll
        for (int v = 0; v < 8; ++v) { best[p][v] = 3.0e38f; bidx[p][v] = 0; }

    // kick off DMA of chunk 0
    prefetch_chunk(lds_off(&sBuf[0][0]), (const char*)ebf, tid);

    // ---- sweep the codebook: 16 chunks x 8 tiles x (16 ds-B reads, 16 WMMA)
#pragma unroll 1
    for (int ck = 0; ck < NCHUNK; ++ck) {
        const int cur = ck & 1;
        wait_async0();        // my async stores to sBuf[cur] are done
        __syncthreads();      // everyone's are; and sBuf[cur^1] is free again

        if (ck + 1 < NCHUNK)
            prefetch_chunk(lds_off(&sBuf[cur ^ 1][0]),
                           (const char*)ebf + (size_t)(ck + 1) * CHUNK_BYTES, tid);

#pragma unroll 1
        for (int t8 = 0; t8 < CHUNK / 16; ++t8) {
            const int rloc = t8 * 16 + m;              // code row in chunk
            const int code = ck * CHUNK + rloc;
            const float en = sEnorm[code];
            // B in the same contiguous k order as A: lane L holds column L%16 (= code),
            // contiguous K-run of 16 at (L/16)*16 -> aligned 32-byte LDS slab.
            const __bf16* bptr = &sBuf[cur][(size_t)rloc * DDIM + hi * 16];

            v8f acc0 = {};
            v8f acc1 = {};
#pragma unroll
            for (int c = 0; c < 8; ++c) {
                v16bf bv = *(const v16bf*)(bptr + c * 32);
                acc0 = __builtin_amdgcn_wmma_f32_16x16x32_bf16(
                           false, a[0][c], false, bv, (short)0, acc0, false, false);
                acc1 = __builtin_amdgcn_wmma_f32_16x16x32_bf16(
                           false, a[1][c], false, bv, (short)0, acc1, false, false);
            }
            // C/D layout: lane L holds col n = L%16; VGPR v holds row v+8*(L/16)
#pragma unroll
            for (int v = 0; v < 8; ++v) {
                const float s0 = en - 2.0f * acc0[v];
                if (s0 < best[0][v]) { best[0][v] = s0; bidx[0][v] = code; }
                const float s1 = en - 2.0f * acc1[v];
                if (s1 < best[1][v]) { best[1][v] = s1; bidx[1][v] = code; }
            }
        }
    }

    // ---- argmin reduce across the 16 columns (within each lane half) ----
#pragma unroll
    for (int p = 0; p < 2; ++p)
#pragma unroll
    for (int v = 0; v < 8; ++v) {
        float d  = best[p][v];
        int   ix = bidx[p][v];
#pragma unroll
        for (int off = 1; off < 16; off <<= 1) {
            const float d2 = __shfl_xor(d, off, 32);
            const int   i2 = __shfl_xor(ix, off, 32);
            if (d2 < d || (d2 == d && i2 < ix)) { d = d2; ix = i2; }
        }
        best[p][v] = d; bidx[p][v] = ix;
    }

    // lanes 0 and 16 hold winners for local rows p*16 + v + 8*hi
    if (m == 0) {
#pragma unroll
        for (int p = 0; p < 2; ++p)
#pragma unroll
        for (int v = 0; v < 8; ++v) {
            const int r  = p * 16 + v + 8 * hi;
            const int ix = bidx[p][v];
            sIdx[waveId][r] = ix;                       // wave-private slab
            out[OFF_IDX + (size_t)(waveRow + r)] = (float)ix;
            atomicAdd(&counts[ix], 1.0f);
        }
    }
    // sIdx is produced and consumed by the same wave; DS ops are in-order.

    // ---- exact f32 gather of winning codebook rows + commitment loss ----
    float lsum = 0.0f;
#pragma unroll 2
    for (int r = 0; r < ROWS_PER_WAVE; ++r) {
        const int ix = sIdx[waveId][r];
        const float* er = embed + (size_t)ix * DDIM;
        const float* zr = z + (size_t)(waveRow + r) * DDIM;
        float*       orow = out + OFF_ZQ + (size_t)(waveRow + r) * DDIM;
#pragma unroll
        for (int j = 0; j < 2; ++j) {
            const int dd = lane * 8 + j * 4;
            float4 e4 = *(const float4*)(er + dd);
            float4 z4 = *(const float4*)(zr + dd);
            *(float4*)(orow + dd) = e4;       // z_q_st forward value == z_q
            const float dx = e4.x - z4.x, dy = e4.y - z4.y;
            const float dz = e4.z - z4.z, dw = e4.w - z4.w;
            lsum += dx * dx + dy * dy + dz * dz + dw * dw;
        }
    }
#pragma unroll
    for (int off = 1; off < 32; off <<= 1) lsum += __shfl_xor(lsum, off, 32);
    if (lane == 0) atomicAdd(lossAcc, lsum);
}

// ---------------------------------------------------------------------------
// Kernel C: scalars — loss scale and perplexity from the histogram.
// ---------------------------------------------------------------------------
__global__ __launch_bounds__(32) void vq_final(const float* __restrict__ counts,
                                               const float* __restrict__ lossAcc,
                                               float* __restrict__ out) {
    const int lane = threadIdx.x;
    float h = 0.0f;
    for (int k = lane; k < KCODES; k += 32) {
        const float p = counts[k] * (1.0f / (float)NROWS);
        h -= p * logf(p + 1e-10f);
    }
#pragma unroll
    for (int off = 1; off < 32; off <<= 1) h += __shfl_xor(h, off, 32);
    if (lane == 0) {
        out[OFF_LOSS] = BETA * lossAcc[0] * (1.0f / ((float)NROWS * (float)DDIM));
        out[OFF_PERP] = expf(h);
    }
}

// ---------------------------------------------------------------------------
extern "C" void kernel_launch(void* const* d_in, const int* in_sizes, int n_in,
                              void* d_out, int out_size, void* d_ws, size_t ws_size,
                              hipStream_t stream) {
    (void)in_sizes; (void)n_in; (void)out_size; (void)ws_size;
    const float* z     = (const float*)d_in[0];   // [16,4096,256] f32
    const float* embed = (const float*)d_in[1];   // [2048,256] f32
    float*       out   = (float*)d_out;

    char* ws = (char*)d_ws;
    __bf16* ebf    = (__bf16*)ws;                                   // 1 MB
    float*  enorm  = (float*)(ws + (size_t)KCODES * DDIM * 2);      // 8 KB
    float*  counts = enorm + KCODES;                                // 8 KB
    float*  lossAc = counts + KCODES;                               // 4 B

    vq_prep <<<KCODES, 32, 0, stream>>>(embed, ebf, enorm, counts, lossAc);
    vq_main <<<NROWS / ROWS_PER_BLOCK, BLOCK_THREADS, 0, stream>>>(
        z, embed, ebf, enorm, out, counts, lossAc);
    vq_final<<<1, 32, 0, stream>>>(counts, lossAc, out);
}
